// AmharicHNet300M_63917703299653
// MI455X (gfx1250) — matrix operand order repaired and checked by reading, not compiled, not run on hardware
//
#include <hip/hip_runtime.h>
#include <hip/hip_bf16.h>
#include <math.h>

// ---------------------------------------------------------------------------
// Model constants
// ---------------------------------------------------------------------------
#define BB   4
#define SS   1024
#define DD   1536
#define HH   12
#define HD   128
#define MAXC 256
#define SM1  1023           // S - 1
#define M1   (BB * SM1)     // 4092 detector rows
#define QROW (3 * DD)       // 4608

typedef __attribute__((ext_vector_type(8)))  float  v8f;
typedef __attribute__((ext_vector_type(16))) __bf16 v16bf;
typedef __bf16 bf16_t;

// ---------------------------------------------------------------------------
// Device helpers
// ---------------------------------------------------------------------------
__device__ __forceinline__ float gelu_f(float v) {
  return 0.5f * v * (1.0f + erff(v * 0.70710678118654752f));
}
__device__ __forceinline__ float wave_sum(float v) {
  for (int o = 16; o > 0; o >>= 1) v += __shfl_xor(v, o, 32);
  return v;
}
__device__ __forceinline__ float wave_max(float v) {
  for (int o = 16; o > 0; o >>= 1) v = fmaxf(v, __shfl_xor(v, o, 32));
  return v;
}

// A-fragment (16x32 bf16, MxK) from an LDS tile.
__device__ __forceinline__ v16bf frag_a_lds(const bf16_t* tile, int stride, int lane) {
  const bf16_t* rowp = tile + (size_t)(lane & 15) * stride;
  const int half = (lane >> 4) * 16;
  v16bf f;
#pragma unroll
  for (int e = 0; e < 16; ++e) {
    const int j = e >> 1, sub = e & 1;
    const int k = half + 2 * j + sub;
    f[e] = rowp[k];
  }
  return f;
}
// B-fragment (32x16 bf16, KxN) sourced from W stored row-major as W[n][k].
__device__ __forceinline__ v16bf frag_b_lds(const bf16_t* tile, int stride, int lane) {
  const bf16_t* rowp = tile + (size_t)(lane & 15) * stride;  // row = n
  const int half = (lane >> 4) * 16;
  v16bf f;
#pragma unroll
  for (int e = 0; e < 16; ++e) {
    const int j = e >> 1, sub = e & 1;
    f[e] = rowp[half + 2 * j + sub];
  }
  return f;
}

// ---------------------------------------------------------------------------
// f32 -> bf16 conversion (grid-stride)
// ---------------------------------------------------------------------------
__global__ void cvt_bf16_kernel(const float* __restrict__ src, bf16_t* __restrict__ dst,
                                size_t n) {
  size_t i = (size_t)blockIdx.x * blockDim.x + threadIdx.x;
  const size_t stride = (size_t)gridDim.x * blockDim.x;
  for (; i < n; i += stride) dst[i] = (bf16_t)src[i];
}

// ---------------------------------------------------------------------------
// Tiled WMMA GEMM:  C[M,N] = act(A[M,K](bf16, row stride lda) @ W[N,K]^T + bias)
// BM=128, BN=64, BK=64, 256 threads (8 waves), wave tile 32x32 (2x2 WMMA tiles)
// 8 WMMAs per barrier stage; b128 staging; prefetch of next K-tile.
// ---------------------------------------------------------------------------
#define GBM 128
#define GBN 64
#define GBK 64
#define GLD (GBK + 8)   // 72 bf16 = 144 B row pitch (16B aligned)

__global__ void __launch_bounds__(256)
gemm_bias_act_kernel(const bf16_t* __restrict__ A, int lda,
                     const bf16_t* __restrict__ W,
                     const float* __restrict__ bias,
                     float* __restrict__ C, bf16_t* __restrict__ Cb,
                     int M, int N, int K, int act) {
  __shared__ bf16_t As[GBM][GLD];
  __shared__ bf16_t Ws[GBN][GLD];

  const int tid  = threadIdx.x;
  const int lane = tid & 31;
  const int w    = tid >> 5;
  const int wm   = (w >> 1) * 32;      // 0,32,64,96
  const int wn   = (w & 1) * 32;       // 0,32
  const int m_block = blockIdx.y * GBM;
  const int n_block = blockIdx.x * GBN;

  v8f acc[2][2] = {};

  for (int k0 = 0; k0 < K; k0 += GBK) {
    // Stage A tile, 16B granules (zero-fill rows past M). 1024 chunks / 256 thr.
    for (int idx = tid; idx < GBM * (GBK / 8); idx += 256) {
      const int r = idx >> 3, c = idx & 7;
      const int gm = m_block + r;
      uint4 val = make_uint4(0u, 0u, 0u, 0u);
      if (gm < M) val = *(const uint4*)(A + (size_t)gm * lda + k0 + c * 8);
      *(uint4*)&As[r][c * 8] = val;
    }
    // Stage W tile (N always a multiple of 64 here). 512 chunks / 256 thr.
    for (int idx = tid; idx < GBN * (GBK / 8); idx += 256) {
      const int r = idx >> 3, c = idx & 7;
      *(uint4*)&Ws[r][c * 8] =
          *(const uint4*)(W + (size_t)(n_block + r) * K + k0 + c * 8);
    }
    // Prefetch next K-tile while this one is consumed.
    if (k0 + GBK < K) {
      const int pr = tid >> 1, pc = (tid & 1) * 32;
      const int gm = m_block + pr;
      if (gm < M) __builtin_prefetch(A + (size_t)gm * lda + k0 + GBK + pc, 0, 0);
      if (tid < GBN)
        __builtin_prefetch(W + (size_t)(n_block + tid) * K + k0 + GBK, 0, 0);
    }
    __syncthreads();

#pragma unroll
    for (int kk = 0; kk < GBK; kk += 32) {
      v16bf af[2], bfr[2];
#pragma unroll
      for (int i = 0; i < 2; ++i) af[i] = frag_a_lds(&As[wm + i * 16][kk], GLD, lane);
#pragma unroll
      for (int j = 0; j < 2; ++j) bfr[j] = frag_b_lds(&Ws[wn + j * 16][kk], GLD, lane);
#pragma unroll
      for (int i = 0; i < 2; ++i)
#pragma unroll
        for (int j = 0; j < 2; ++j)
          acc[i][j] = __builtin_amdgcn_wmma_f32_16x16x32_bf16(
              false, af[i], false, bfr[j], (short)0, acc[i][j], false, false);
    }
    __syncthreads();
  }

  const int nlane = lane & 15;
  const int mhalf = (lane >> 4) * 8;
#pragma unroll
  for (int i = 0; i < 2; ++i)
#pragma unroll
    for (int j = 0; j < 2; ++j) {
      const int gn = n_block + wn + j * 16 + nlane;
      const float bv = bias ? bias[gn] : 0.0f;
#pragma unroll
      for (int r = 0; r < 8; ++r) {
        const int gm = m_block + wm + i * 16 + mhalf + r;
        if (gm < M) {
          float v = acc[i][j][r] + bv;
          if (act == 1) v = gelu_f(v);
          const size_t off = (size_t)gm * N + gn;
          if (C)  C[off]  = v;
          if (Cb) Cb[off] = (bf16_t)v;
        }
      }
    }
}

// ---------------------------------------------------------------------------
// bi[b,s,:] = concat(x[b,s], x[b,s+1]) as bf16  (contiguous 3072 floats in x)
// ---------------------------------------------------------------------------
__global__ void build_bi_kernel(const float* __restrict__ x, bf16_t* __restrict__ bi) {
  const int bs = blockIdx.x;               // [0, B*1023)
  const int b = bs / SM1, s = bs % SM1;
  const float* x0 = x + ((size_t)b * SS + s) * DD;
  bf16_t* o = bi + (size_t)bs * (2 * DD);
  for (int k = threadIdx.x; k < 2 * DD; k += blockDim.x) o[k] = (bf16_t)x0[k];
}

// ---------------------------------------------------------------------------
// Cosine similarity between consecutive (strided) x_ling rows. 1 wave / pair.
// ---------------------------------------------------------------------------
__global__ void cos_kernel(const float* __restrict__ xling, float* __restrict__ cs,
                           int Lp, int scaleS) {
  const int wid  = blockIdx.x * (blockDim.x >> 5) + (threadIdx.x >> 5);
  const int lane = threadIdx.x & 31;
  if (wid >= BB * Lp) return;
  const int b = wid / Lp, j = wid % Lp;
  const float* pa = xling + ((size_t)b * SS + (size_t)j * scaleS) * DD;
  const float* pb = xling + ((size_t)b * SS + (size_t)(j + 1) * scaleS) * DD;
  float dot = 0.f, na = 0.f, nb = 0.f;
  for (int d = lane; d < DD; d += 32) {
    const float av = pa[d], bv = pb[d];
    dot += av * bv; na += av * av; nb += bv * bv;
  }
  dot = wave_sum(dot); na = wave_sum(na); nb = wave_sum(nb);
  if (lane == 0)
    cs[wid] = dot / (fmaxf(sqrtf(na), 1e-8f) * fmaxf(sqrtf(nb), 1e-8f));
}

__device__ __forceinline__ float interp1(const float* y, int Lin, int t) {
  float src = ((float)t + 0.5f) * ((float)Lin / (float)SM1) - 0.5f;
  src = fminf(fmaxf(src, 0.0f), (float)(Lin - 1));
  const int i0 = (int)floorf(src);
  const int i1 = (i0 + 1 < Lin - 1) ? (i0 + 1) : (Lin - 1);
  const float w = src - (float)i0;
  return y[i0] * (1.0f - w) + y[i1] * w;
}

__global__ void base_kernel(const float* __restrict__ cs1, const float* __restrict__ cs2,
                            const float* __restrict__ cs4, float* __restrict__ base) {
  const int b = blockIdx.x;
  for (int t = threadIdx.x; t < SM1; t += blockDim.x) {
    const float v1 = cs1[b * SM1 + t];                 // identity interp (Lin==Lout)
    const float v2 = interp1(cs2 + b * 511, 511, t);
    const float v4 = interp1(cs4 + b * 255, 255, t);
    base[b * SM1 + t] = 0.5f * (1.0f - (v1 + v2 + v4) * (1.0f / 3.0f));
  }
}

// ---------------------------------------------------------------------------
// Detector head: learned[n,ms] = sigmoid(h2[n,ms,:] . detW3[n] + detb3[n])
// ---------------------------------------------------------------------------
__global__ void det3_kernel(const float* __restrict__ h2, const float* __restrict__ detW3,
                            const float* __restrict__ detb3, float* __restrict__ learned) {
  const int wid  = blockIdx.x * (blockDim.x >> 5) + (threadIdx.x >> 5);
  const int lane = threadIdx.x & 31;
  if (wid >= 3 * M1) return;
  const int n = wid / M1, ms = wid % M1;
  const float* hp = h2 + ((size_t)n * M1 + ms) * (DD / 2);
  const float* wp = detW3 + (size_t)n * (DD / 2);
  float dot = 0.f;
  for (int d = lane; d < DD / 2; d += 32) dot += hp[d] * wp[d];
  dot = wave_sum(dot);
  if (lane == 0)
    learned[(size_t)n * M1 + ms] = 1.0f / (1.0f + __expf(-(dot + detb3[n])));
}

// ---------------------------------------------------------------------------
// final = 0.6*base + 0.4*mean(learned); boundaries>0.5; seg = cumsum - 1
// ---------------------------------------------------------------------------
__global__ void final_seg_kernel(const float* __restrict__ base,
                                 const float* __restrict__ learned,
                                 int* __restrict__ seg) {
  __shared__ int flags[SM1];
  const int b = blockIdx.x;
  for (int t = threadIdx.x; t < SM1; t += blockDim.x) {
    const int ms = b * SM1 + t;
    const float lm = (learned[ms] + learned[M1 + ms] + learned[2 * M1 + ms]) * (1.0f / 3.0f);
    const float fin = 0.6f * base[ms] + 0.4f * lm;
    flags[t] = (fin > 0.5f) ? 1 : 0;
  }
  __syncthreads();
  if (threadIdx.x == 0) {
    int run = 0;
    seg[b * SS] = 0;
    for (int t = 1; t < SS; ++t) { run += flags[t - 1]; seg[b * SS + t] = run; }
  }
}

// ---------------------------------------------------------------------------
// Fused segment-masked attention. 1 workgroup per (b, h, 16-row q-tile).
// Scores live in dynamic LDS (16 x 1024 f32); WMMA for QK^T and P@V.
// ---------------------------------------------------------------------------
#define ATTN_SMEM (16 * SS * 4 + 16 * (HD + 8) * 2 + 16 * 4)

__global__ void __launch_bounds__(256)
attn_kernel(const bf16_t* __restrict__ qkv, const int* __restrict__ seg,
            bf16_t* __restrict__ ctx) {
  extern __shared__ unsigned char smem[];
  float*  Sc   = (float*)smem;                                   // [16][1024]
  bf16_t* Qs   = (bf16_t*)(smem + 16 * SS * 4);                  // [16][HD+8]
  int*    SegQ = (int*)(smem + 16 * SS * 4 + 16 * (HD + 8) * 2); // [16]

  const int tid = threadIdx.x, lane = tid & 31, w = tid >> 5;
  const int bid = blockIdx.x;
  const int qt = bid & 63;
  const int h  = (bid >> 6) % HH;
  const int b  = bid / (64 * HH);
  const int q0 = qt * 16;

  for (int idx = tid; idx < 16 * HD; idx += 256) {
    const int r = idx >> 7, c = idx & (HD - 1);
    Qs[r * (HD + 8) + c] = qkv[((size_t)b * SS + q0 + r) * QROW + h * HD + c];
  }
  if (tid < 16) SegQ[tid] = seg[b * SS + q0 + tid];
  __syncthreads();

  // ---- scores = (Q K^T) * scale, masked by segment ----
  const float scale = 0.08838834764831845f;  // 1/sqrt(128)
  for (int ct = w; ct < SS / 16; ct += 8) {
    const int c0 = ct * 16;
    v8f acc = {};
#pragma unroll
    for (int ks = 0; ks < HD; ks += 32) {
      const v16bf a = frag_a_lds(Qs + ks, HD + 8, lane);
      v16bf bb;
      {
        const int n = lane & 15, half = (lane >> 4) * 16;
        const bf16_t* kp = qkv + ((size_t)b * SS + c0 + n) * QROW + DD + h * HD + ks;
#pragma unroll
        for (int e = 0; e < 16; ++e) {
          const int j = e >> 1, sub = e & 1;
          bb[e] = kp[half + 2 * j + sub];
        }
      }
      acc = __builtin_amdgcn_wmma_f32_16x16x32_bf16(false, a, false, bb, (short)0,
                                                    acc, false, false);
    }
    const int n = lane & 15, mh = (lane >> 4) * 8;
    const int cs = c0 + n;
    const int sgc = seg[b * SS + cs];
#pragma unroll
    for (int r = 0; r < 8; ++r) {
      const int m = mh + r;
      Sc[m * SS + cs] = (SegQ[m] == sgc) ? acc[r] * scale : -1.0e9f;
    }
  }
  __syncthreads();

  // ---- softmax: wave w handles rows 2w, 2w+1 ----
#pragma unroll
  for (int rr = 0; rr < 2; ++rr) {
    float* row = Sc + (w * 2 + rr) * SS;
    float mx = -3.4e38f;
    for (int c = lane; c < SS; c += 32) mx = fmaxf(mx, row[c]);
    mx = wave_max(mx);
    float sum = 0.f;
    for (int c = lane; c < SS; c += 32) {
      const float e = __expf(row[c] - mx);
      row[c] = e; sum += e;
    }
    sum = wave_sum(sum);
    const float inv = 1.0f / sum;
    for (int c = lane; c < SS; c += 32) row[c] *= inv;
  }
  __syncthreads();

  // ---- ctx = P @ V ; wave w owns HD columns [16w, 16w+16) ----
  const int n0 = w * 16;
  v8f acc = {};
  for (int k0 = 0; k0 < SS; k0 += 32) {
    v16bf a;
    {
      const int mr = lane & 15, half = (lane >> 4) * 16;
      const float* rp = Sc + mr * SS + k0;
#pragma unroll
      for (int e = 0; e < 16; ++e) {
        const int j = e >> 1, sub = e & 1;
        const int k = half + 2 * j + sub;
        a[e] = (bf16_t)rp[k];
      }
    }
    v16bf bb;
    {
      const int n = lane & 15, half = (lane >> 4) * 16;
      const bf16_t* vp = qkv + ((size_t)b * SS + k0) * QROW + 2 * DD + h * HD + n0 + n;
#pragma unroll
      for (int e = 0; e < 16; ++e) {
        const int j = e >> 1, sub = e & 1;
        bb[e] = vp[(size_t)(half + 2 * j + sub) * QROW];
      }
    }
    acc = __builtin_amdgcn_wmma_f32_16x16x32_bf16(false, a, false, bb, (short)0,
                                                  acc, false, false);
  }
  const int n = lane & 15, mh = (lane >> 4) * 8;
#pragma unroll
  for (int r = 0; r < 8; ++r)
    ctx[((size_t)b * SS + q0 + mh + r) * DD + h * HD + n0 + n] = (bf16_t)acc[r];
}

// ---------------------------------------------------------------------------
// Segment pooling
// ---------------------------------------------------------------------------
__global__ void zero_pool_kernel(float* __restrict__ sums, int* __restrict__ counts) {
  const size_t n = (size_t)BB * MAXC * DD;
  const size_t stride = (size_t)gridDim.x * blockDim.x;
  for (size_t i = (size_t)blockIdx.x * blockDim.x + threadIdx.x; i < n; i += stride)
    sums[i] = 0.0f;
  const size_t i0 = (size_t)blockIdx.x * blockDim.x + threadIdx.x;
  if (i0 < BB * MAXC) counts[i0] = 0;
}

__global__ void pool_scatter_kernel(const float* __restrict__ attn_out,
                                    const int* __restrict__ seg,
                                    float* __restrict__ sums, int* __restrict__ counts) {
  const int bs = blockIdx.x;               // [0, B*S)
  const int b = bs >> 10, s = bs & (SS - 1);
  const int sg = seg[b * SS + s];
  if (sg >= MAXC) return;                  // dropped by one-hot in reference
  if (threadIdx.x == 0) atomicAdd(&counts[b * MAXC + sg], 1);
  const float* src = attn_out + (size_t)bs * DD;
  float* dst = sums + ((size_t)b * MAXC + sg) * DD;
  for (int d = threadIdx.x; d < DD; d += blockDim.x) atomicAdd(&dst[d], src[d]);
}

__global__ void chunk_make_kernel(const float* __restrict__ sums,
                                  const int* __restrict__ counts,
                                  const float* __restrict__ size_emb,
                                  const float* __restrict__ pos_enc,
                                  bf16_t* __restrict__ chunk_bf) {
  const int bm = blockIdx.x;               // [0, B*MAXC)
  const int m = bm & (MAXC - 1);
  const int cnt = counts[bm];
  const float denom = fmaxf((float)cnt, 1.0f);
  const int clen = (cnt < 1023) ? cnt : 1023;  // MAXLEN-1
  for (int d = threadIdx.x; d < DD; d += blockDim.x) {
    float v = 0.0f;
    if (cnt > 0) v = sums[(size_t)bm * DD + d] / denom + size_emb[(size_t)clen * DD + d];
    v += pos_enc[(size_t)m * DD + d];
    chunk_bf[(size_t)bm * DD + d] = (bf16_t)v;
  }
}

// ---------------------------------------------------------------------------
// LayerNorm rows of 1536 -> d_out
// ---------------------------------------------------------------------------
__global__ void layernorm_kernel(const float* __restrict__ y, const float* __restrict__ g,
                                 const float* __restrict__ bb, float* __restrict__ out) {
  __shared__ float rs[256], rs2[256];
  const int row = blockIdx.x;
  const float* yp = y + (size_t)row * DD;
  float s = 0.f, s2 = 0.f;
  for (int d = threadIdx.x; d < DD; d += 256) {
    const float v = yp[d]; s += v; s2 += v * v;
  }
  rs[threadIdx.x] = s; rs2[threadIdx.x] = s2;
  __syncthreads();
  for (int o = 128; o > 0; o >>= 1) {
    if (threadIdx.x < o) { rs[threadIdx.x] += rs[threadIdx.x + o];
                           rs2[threadIdx.x] += rs2[threadIdx.x + o]; }
    __syncthreads();
  }
  const float mu  = rs[0] * (1.0f / DD);
  const float var = rs2[0] * (1.0f / DD) - mu * mu;
  const float inv = rsqrtf(var + 1e-5f);
  for (int d = threadIdx.x; d < DD; d += 256)
    out[(size_t)row * DD + d] = (yp[d] - mu) * inv * g[d] + bb[d];
}

// ---------------------------------------------------------------------------
// Host side
// ---------------------------------------------------------------------------
static void launch_gemm(const bf16_t* A, int lda, const bf16_t* W, const float* bias,
                        float* C, bf16_t* Cb, int M, int N, int K, int act,
                        hipStream_t s) {
  dim3 grid(N / GBN, (M + GBM - 1) / GBM);
  gemm_bias_act_kernel<<<grid, 256, 0, s>>>(A, lda, W, bias, C, Cb, M, N, K, act);
}

static void launch_cvt(const float* src, bf16_t* dst, size_t n, hipStream_t s) {
  int blocks = (int)((n + 255) / 256);
  if (blocks > 2048) blocks = 2048;
  cvt_bf16_kernel<<<blocks, 256, 0, s>>>(src, dst, n);
}

extern "C" void kernel_launch(void* const* d_in, const int* in_sizes, int n_in,
                              void* d_out, int out_size, void* d_ws, size_t ws_size,
                              hipStream_t stream) {
  const float* x         = (const float*)d_in[0];
  const float* Wp        = (const float*)d_in[1];
  const float* bp        = (const float*)d_in[2];
  const float* detW1     = (const float*)d_in[3];
  const float* detb1     = (const float*)d_in[4];
  const float* detW2     = (const float*)d_in[5];
  const float* detb2     = (const float*)d_in[6];
  const float* detW3     = (const float*)d_in[7];
  const float* detb3     = (const float*)d_in[8];
  const float* in_proj_w = (const float*)d_in[9];
  const float* in_proj_b = (const float*)d_in[10];
  const float* out_w     = (const float*)d_in[11];
  const float* out_b     = (const float*)d_in[12];
  const float* size_emb  = (const float*)d_in[13];
  const float* pos_enc   = (const float*)d_in[14];
  const float* procW1    = (const float*)d_in[15];
  const float* procb1    = (const float*)d_in[16];
  const float* procW2    = (const float*)d_in[17];
  const float* procb2    = (const float*)d_in[18];
  const float* ln_g      = (const float*)d_in[19];
  const float* ln_b      = (const float*)d_in[20];

  char* ws = (char*)d_ws;
  size_t cur = 0;
  auto alloc = [&](size_t bytes) -> char* {
    char* p = ws + cur;
    cur += (bytes + 255) & ~(size_t)255;
    return p;
  };

  bf16_t* x_bf     = (bf16_t*)alloc((size_t)BB * SS * DD * 2);
  bf16_t* wp_bf    = (bf16_t*)alloc((size_t)DD * DD * 2);
  bf16_t* dw1_bf   = (bf16_t*)alloc((size_t)3 * DD * 2 * DD * 2);
  bf16_t* dw2_bf   = (bf16_t*)alloc((size_t)3 * (DD / 2) * DD * 2);
  bf16_t* ipw_bf   = (bf16_t*)alloc((size_t)3 * DD * DD * 2);
  bf16_t* ow_bf    = (bf16_t*)alloc((size_t)DD * DD * 2);
  bf16_t* pw1_bf   = (bf16_t*)alloc((size_t)2 * DD * DD * 2);
  bf16_t* pw2_bf   = (bf16_t*)alloc((size_t)DD * 2 * DD * 2);
  float*  xling    = (float*)alloc((size_t)BB * SS * DD * 4);
  float*  cs1      = (float*)alloc((size_t)BB * SM1 * 4);
  float*  cs2      = (float*)alloc((size_t)BB * 511 * 4);
  float*  cs4      = (float*)alloc((size_t)BB * 255 * 4);
  float*  base     = (float*)alloc((size_t)BB * SM1 * 4);
  bf16_t* bi_bf    = (bf16_t*)alloc((size_t)M1 * 2 * DD * 2);
  bf16_t* h1_bf    = (bf16_t*)alloc((size_t)M1 * 3 * DD * 2);
  float*  h2       = (float*)alloc((size_t)3 * M1 * (DD / 2) * 4);
  float*  learned  = (float*)alloc((size_t)3 * M1 * 4);
  int*    seg      = (int*)alloc((size_t)BB * SS * 4);
  bf16_t* qkv_bf   = (bf16_t*)alloc((size_t)BB * SS * 3 * DD * 2);
  bf16_t* ctx_bf   = (bf16_t*)alloc((size_t)BB * SS * DD * 2);
  float*  attn_out = (float*)alloc((size_t)BB * SS * DD * 4);
  float*  sums     = (float*)alloc((size_t)BB * MAXC * DD * 4);
  int*    counts   = (int*)alloc((size_t)BB * MAXC * 4);
  bf16_t* chunk_bf = (bf16_t*)alloc((size_t)BB * MAXC * DD * 2);
  bf16_t* hproc_bf = (bf16_t*)alloc((size_t)BB * MAXC * 2 * DD * 2);
  float*  ybuf     = (float*)alloc((size_t)BB * MAXC * DD * 4);

  // 0) bf16 conversions (one-time per call)
  launch_cvt(x, x_bf, (size_t)BB * SS * DD, stream);
  launch_cvt(Wp, wp_bf, (size_t)DD * DD, stream);
  launch_cvt(detW1, dw1_bf, (size_t)3 * DD * 2 * DD, stream);
  launch_cvt(detW2, dw2_bf, (size_t)3 * (DD / 2) * DD, stream);
  launch_cvt(in_proj_w, ipw_bf, (size_t)3 * DD * DD, stream);
  launch_cvt(out_w, ow_bf, (size_t)DD * DD, stream);
  launch_cvt(procW1, pw1_bf, (size_t)2 * DD * DD, stream);
  launch_cvt(procW2, pw2_bf, (size_t)DD * 2 * DD, stream);

  // 1) x_ling = x @ Wp^T + bp
  launch_gemm(x_bf, DD, wp_bf, bp, xling, nullptr, BB * SS, DD, DD, 0, stream);

  // 2) multi-scale cosine similarity + base boundary prob
  {
    int Lp = SM1;  cos_kernel<<<(BB * Lp + 7) / 8, 256, 0, stream>>>(xling, cs1, Lp, 1);
    Lp = 511;      cos_kernel<<<(BB * Lp + 7) / 8, 256, 0, stream>>>(xling, cs2, Lp, 2);
    Lp = 255;      cos_kernel<<<(BB * Lp + 7) / 8, 256, 0, stream>>>(xling, cs4, Lp, 4);
    base_kernel<<<BB, 256, 0, stream>>>(cs1, cs2, cs4, base);
  }

  // 3) detector MLP (n folded into N for layer 1)
  build_bi_kernel<<<M1, 256, 0, stream>>>(x, bi_bf);
  launch_gemm(bi_bf, 2 * DD, dw1_bf, detb1, nullptr, h1_bf, M1, 3 * DD, 2 * DD, 1, stream);
  for (int n = 0; n < 3; ++n)
    launch_gemm(h1_bf + (size_t)n * DD, 3 * DD,
                dw2_bf + (size_t)n * (DD / 2) * DD, detb2 + (size_t)n * (DD / 2),
                h2 + (size_t)n * M1 * (DD / 2), nullptr, M1, DD / 2, DD, 1, stream);
  det3_kernel<<<(3 * M1 + 7) / 8, 256, 0, stream>>>(h2, detW3, detb3, learned);
  final_seg_kernel<<<BB, 256, 0, stream>>>(base, learned, seg);

  // 4) qkv projection, fused segment-masked attention, output projection
  launch_gemm(x_bf, DD, ipw_bf, in_proj_b, nullptr, qkv_bf, BB * SS, 3 * DD, DD, 0, stream);
  (void)hipFuncSetAttribute((const void*)attn_kernel,
                            hipFuncAttributeMaxDynamicSharedMemorySize, ATTN_SMEM);
  attn_kernel<<<BB * HH * (SS / 16), 256, ATTN_SMEM, stream>>>(qkv_bf, seg, ctx_bf);
  launch_gemm(ctx_bf, DD, ow_bf, out_b, attn_out, nullptr, BB * SS, DD, DD, 0, stream);

  // 5) segment pooling -> chunk embedding
  zero_pool_kernel<<<2048, 256, 0, stream>>>(sums, counts);
  pool_scatter_kernel<<<BB * SS, 256, 0, stream>>>(attn_out, seg, sums, counts);
  chunk_make_kernel<<<BB * MAXC, 256, 0, stream>>>(sums, counts, size_emb, pos_enc, chunk_bf);

  // 6) processor MLP + LayerNorm -> d_out
  launch_gemm(chunk_bf, DD, pw1_bf, procb1, nullptr, hproc_bf, BB * MAXC, 2 * DD, DD, 1, stream);
  launch_gemm(hproc_bf, 2 * DD, pw2_bf, procb2, ybuf, nullptr, BB * MAXC, DD, 2 * DD, 0, stream);
  layernorm_kernel<<<BB * MAXC, 256, 0, stream>>>(ybuf, ln_g, ln_b, (float*)d_out);
}
